// Arcface_MV_88192858456488
// MI455X (gfx1250) — matrix operand order repaired and checked by reading, not compiled, not run on hardware
//
#include <hip/hip_runtime.h>
#include <hip/hip_bf16.h>
#include <math.h>

// ---------------- problem constants ----------------
#define NB    512
#define EMB   512
#define NCLS  51332
#define SCALE 64.0f
#define T_MV  0.2f
#define COS_M 0.8775825618903728f   // cos(0.5)
#define SIN_M 0.4794255386042030f   // sin(0.5)

// ---------------- GEMM tiling ----------------------
#define BM 128
#define BN 128
#define BK 32
#define KCH (EMB / BK)                 // 16 K-chunks
#define BPITCH 40                      // LDS row pitch (ushorts): 80B, 16B-aligned frags
#define NUMBC ((NCLS + BN - 1) / BN)   // 402 column-blocks

typedef __attribute__((ext_vector_type(16))) unsigned short v16u;
typedef __attribute__((ext_vector_type(8)))  unsigned short v8u;
typedef __attribute__((ext_vector_type(16))) __bf16         v16bf;
typedef __attribute__((ext_vector_type(8)))  float          v8f;

union FragU { v8u half[2]; v16u full; };

__device__ __forceinline__ unsigned short f2bf(float f) {
    // round-to-nearest-even f32 -> bf16 bits (scalar helper, cold paths only)
    unsigned int u = __builtin_bit_cast(unsigned int, f);
    u += 0x7FFFu + ((u >> 16) & 1u);
    return (unsigned short)(u >> 16);
}

// CDNA5 VOP3P mixed-precision FMA-with-convert (ISA 15.10, opc 62/63):
// v_fma_mixlo_bf16: D[15:0]  = bf16(fma(f32 src0, 1.0, 0))
// v_fma_mixhi_bf16: D[31:16] = bf16(fma(f32 src0, 1.0, 0)), low half preserved
// -> converts + packs a bf16 pair in 2 VALU ops.
__device__ __forceinline__ unsigned int pack2bf(float lo, float hi) {
    unsigned int r;
    asm("v_fma_mixlo_bf16 %0, %1, 1.0, 0\n\t"
        "v_fma_mixhi_bf16 %0, %2, 1.0, 0"
        : "=&v"(r)
        : "v"(lo), "v"(hi));
    return r;
}

// ---------------------------------------------------------------------------
__global__ void arc_zero(float* out) {
    if (threadIdx.x == 0 && blockIdx.x == 0) out[0] = 0.0f;
}

// inv_norm[n] = 1 / ||kernel[:,n]||  (105 MB streaming read, coalesced in n)
__global__ void arc_colnorm(const float* __restrict__ kern,
                            float* __restrict__ inv_norm) {
    int n = blockIdx.x * blockDim.x + threadIdx.x;
    if (n >= NCLS) return;
    float s = 0.0f;
    for (int e = 0; e < EMB; ++e) {
        float v = kern[e * NCLS + n];
        s = fmaf(v, v, s);
    }
    inv_norm[n] = rsqrtf(s);
}

// emb f32 -> bf16, pre-packed into the exact per-lane WMMA A-fragment layout:
// Apack[((mt*KCH + kc)*32 + lane)*16 + e]  (each lane's 16 values contiguous)
__global__ void arc_packA(const float* __restrict__ emb,
                          unsigned short* __restrict__ Apack) {
    int o = blockIdx.x * blockDim.x + threadIdx.x;
    if (o >= NB * EMB) return;
    int e    = o & 15;
    int lane = (o >> 4) & 31;
    int kc   = (o >> 9) & 15;
    int mt   = o >> 13;
    int h = lane >> 4, ln = lane & 15;
    int v = e >> 1, j = e & 1;
    int kk = (v < 4) ? (8 * h + 2 * v) : (16 + 8 * h + 2 * (v - 4));
    int K  = kc * BK + kk + j;
    int row = mt * 16 + ln;
    Apack[o] = f2bf(emb[row * EMB + K]);
}

// per-row ground-truth cosine + margin constants; one wave32 per row
__global__ void arc_gt(const float* __restrict__ emb,
                       const float* __restrict__ kern,
                       const int*   __restrict__ label,
                       const float* __restrict__ inv_norm,
                       float* __restrict__ gt_cos,
                       float* __restrict__ final_gt,
                       float* __restrict__ x_label) {
    int wid  = threadIdx.x >> 5;
    int lane = threadIdx.x & 31;
    int r = blockIdx.x * 8 + wid;
    if (r >= NB) return;
    int lab = label[r];
    float p = 0.0f;
    for (int e = lane; e < EMB; e += 32)
        p = fmaf(emb[r * EMB + e], kern[e * NCLS + lab], p);
    for (int off = 16; off >= 1; off >>= 1)
        p += __shfl_xor(p, off, 32);
    if (lane == 0) {
        float gt = fminf(1.0f, fmaxf(-1.0f, p * inv_norm[lab]));
        float st = sqrtf(fmaxf(0.0f, 1.0f - gt * gt));
        float gc = gt * COS_M - st * SIN_M;       // cos(theta + m)
        float fg = (gt > 0.0f) ? gc : gt;          // easy margin
        gt_cos[r]   = gc;
        final_gt[r] = fg;
        x_label[r]  = fg * SCALE;                  // logit at label column
    }
}

// ---------------------------------------------------------------------------
// Fused GEMM + ArcFace margin + per-(row, column-block) softmax partials.
// Block tile 128x128, 8 waves as 4(M) x 2(N); each wave: 32x64 via 8 WMMAs/K-step
// (2 A-tiles x 4 B-tiles -> each B fragment feeds 2 WMMAs, halving LDS reads
// per matrix op). A fragments pre-packed in global (L2-resident). B staged to
// LDS column-major bf16 with double buffering; fragment loads are ds_load_b128.
// Tail columns use a clamped address (garbage ok: epilogue masks them to -inf).
// K-loop force-unrolled so buffer parity & chunk offsets fold into immediates.
__global__ __launch_bounds__(256)
void arc_gemm(const float* __restrict__ kern,
              const unsigned short* __restrict__ Apack,
              const float* __restrict__ inv_norm,
              const float* __restrict__ gt_cos,
              const float* __restrict__ final_gt,
              const int*   __restrict__ label,
              float* __restrict__ partials) {
    __shared__ __align__(16) unsigned short Bs[2][BN][BPITCH];  // 2 x 128 x 40
    __shared__ float redM[8][16][2];
    __shared__ float redS[8][16][2];

    const int tid  = threadIdx.x;
    const int wid  = tid >> 5;
    const int lane = tid & 31;
    const int wm   = wid >> 1;          // 0..3  (M wave: 32 rows each)
    const int wn   = wid & 1;           // 0..1  (N wave: 64 cols each)
    const int h    = lane >> 4;         // lane half
    const int ln   = lane & 15;
    const int mBase = blockIdx.y * BM;
    const int nBase = blockIdx.x * BN;
    const int mta   = blockIdx.y * 8 + wm * 2;   // first 16-row tile of this wave

    // staging decomposition: unit = (K pair, 4 cols); 512 units, 2 per thread
    const int kp0 = (tid) >> 5;          // rep 0: k-pair 0..7
    const int kp1 = (tid + 256) >> 5;    // rep 1: k-pair 8..15
    const int jg  = tid & 31;
    const int j4  = jg << 2;             // col 0,4,...,124
    const int gcol = nBase + j4;
    const int gcc  = (gcol < NCLS - 3) ? gcol : (NCLS - 4);  // clamped, no branch

    v8f acc[2][4] = {{{}, {}, {}, {}}, {{}, {}, {}, {}}};

    // ---- preload K-chunk 0 into Bs[0] (column-major bf16, K-pair packed) ----
    {
        const int kp[2] = {kp0, kp1};
        #pragma unroll
        for (int rep = 0; rep < 2; ++rep) {
            int k = kp[rep] << 1;
            float4 a = *(const float4*)(kern + k * NCLS + gcc);
            float4 b = *(const float4*)(kern + (k + 1) * NCLS + gcc);
            *(unsigned int*)(&Bs[0][j4 + 0][k]) = pack2bf(a.x, b.x);
            *(unsigned int*)(&Bs[0][j4 + 1][k]) = pack2bf(a.y, b.y);
            *(unsigned int*)(&Bs[0][j4 + 2][k]) = pack2bf(a.z, b.z);
            *(unsigned int*)(&Bs[0][j4 + 3][k]) = pack2bf(a.w, b.w);
        }
    }
    __syncthreads();

    #pragma unroll
    for (int kc = 0; kc < KCH; ++kc) {
        const int  cur  = kc & 1;
        const bool have = (kc + 1) < KCH;

        // ---- issue next chunk's global loads early (latency overlap) ----
        float4 pa[2], pb[2];
        if (have) {
            const int kp[2] = {kp0, kp1};
            #pragma unroll
            for (int rep = 0; rep < 2; ++rep) {
                int k = (kc + 1) * BK + (kp[rep] << 1);
                pa[rep] = *(const float4*)(kern + k * NCLS + gcc);
                pb[rep] = *(const float4*)(kern + (k + 1) * NCLS + gcc);
            }
        }
        if (kc + 2 < KCH) {
            __builtin_prefetch(kern + ((kc + 2) * BK + (tid >> 4)) * NCLS + gcc, 0, 1);
        }

        // ---- A fragments: two 32B contiguous per-lane loads (pre-packed) ----
        v16bf af[2];
        #pragma unroll
        for (int a = 0; a < 2; ++a) {
            v16u au = *(const v16u*)(Apack + ((((mta + a) * KCH) + kc) * 32 + lane) * 16);
            af[a] = __builtin_bit_cast(v16bf, au);
        }

        // ---- B fragments + 8 WMMAs (each B fragment feeds both A tiles) ----
        #pragma unroll
        for (int t = 0; t < 4; ++t) {
            const unsigned short* p = &Bs[cur][wn * 64 + t * 16 + ln][8 * h];
            FragU b;
            b.half[0] = *(const v8u*)(p);
            b.half[1] = *(const v8u*)(p + 16);
            v16bf bfv = __builtin_bit_cast(v16bf, b.full);
            acc[0][t] = __builtin_amdgcn_wmma_f32_16x16x32_bf16(
                            false, af[0], false, bfv, (short)0, acc[0][t], false, false);
            acc[1][t] = __builtin_amdgcn_wmma_f32_16x16x32_bf16(
                            false, af[1], false, bfv, (short)0, acc[1][t], false, false);
        }

        // ---- convert + store next chunk into the other buffer ----
        if (have) {
            const int kp[2] = {kp0, kp1};
            #pragma unroll
            for (int rep = 0; rep < 2; ++rep) {
                int k = kp[rep] << 1;
                *(unsigned int*)(&Bs[cur ^ 1][j4 + 0][k]) = pack2bf(pa[rep].x, pb[rep].x);
                *(unsigned int*)(&Bs[cur ^ 1][j4 + 1][k]) = pack2bf(pa[rep].y, pb[rep].y);
                *(unsigned int*)(&Bs[cur ^ 1][j4 + 2][k]) = pack2bf(pa[rep].z, pb[rep].z);
                *(unsigned int*)(&Bs[cur ^ 1][j4 + 3][k]) = pack2bf(pa[rep].w, pb[rep].w);
            }
        }
        __syncthreads();
    }

    // ---- epilogue: margin transform + per-row (max, sum-exp) over this tile ----
    #pragma unroll
    for (int a = 0; a < 2; ++a) {
        const int rowTile = wm * 2 + a;              // 0..7 (16-row tile in block)
        const int grBase  = mBase + rowTile * 16;
        #pragma unroll
        for (int r = 0; r < 8; ++r) {
            int m_local = r + 8 * h;                 // C/D layout: VGPR r, half h
            int gr  = grBase + m_local;
            int lab = label[gr];
            float thr = gt_cos[gr];
            float fg  = final_gt[gr];

            float mt2[4], st2[4];
            #pragma unroll
            for (int t = 0; t < 4; ++t) {
                int gc = nBase + wn * 64 + t * 16 + ln;
                bool valid = (gc < NCLS);
                int gci = valid ? gc : (NCLS - 1);
                float av = acc[a][t][r];
                float iv = inv_norm[gci];
                float c  = fminf(1.0f, fmaxf(-1.0f, av * iv));
                float o  = (c > thr) ? ((T_MV + 1.0f) * c + T_MV) : c;
                if (valid && gc == lab) o = fg;
                float logit = o * SCALE;
                float v = valid ? logit : -1e30f;
                #pragma unroll
                for (int msk = 8; msk >= 1; msk >>= 1)
                    v = fmaxf(v, __shfl_xor(v, msk, 32));
                float e = valid ? __expf(logit - v) : 0.0f;
                #pragma unroll
                for (int msk = 8; msk >= 1; msk >>= 1)
                    e += __shfl_xor(e, msk, 32);
                mt2[t] = v; st2[t] = e;
            }
            float Mr = fmaxf(fmaxf(mt2[0], mt2[1]), fmaxf(mt2[2], mt2[3]));
            float Sr = 0.0f;
            #pragma unroll
            for (int t = 0; t < 4; ++t)
                if (st2[t] > 0.0f) Sr += st2[t] * __expf(mt2[t] - Mr);
            if (ln == 0) { redM[rowTile][m_local][wn] = Mr; redS[rowTile][m_local][wn] = Sr; }
        }
    }
    __syncthreads();

    // combine the two N-waves; one thread per row of the block tile
    if (tid < BM) {
        int wmr = tid >> 4, m = tid & 15;
        float M = -1e30f, S = 0.0f;
        #pragma unroll
        for (int w = 0; w < 2; ++w) {
            float mb = redM[wmr][m][w], sb = redS[wmr][m][w];
            if (sb > 0.0f) {
                float nM = fmaxf(M, mb);
                S = S * __expf(M - nM) + sb * __expf(mb - nM);
                M = nM;
            }
        }
        int gr = mBase + tid;
        int pi = (gr * NUMBC + blockIdx.x) * 2;
        partials[pi]     = M;
        partials[pi + 1] = S;
    }
}

// ---------------------------------------------------------------------------
// online log-sum-exp across the 402 column-block partials; -mean(logp[label])
__global__ void arc_finalize(const float* __restrict__ partials,
                             const float* __restrict__ x_label,
                             float* __restrict__ out) {
    int r = blockIdx.x * blockDim.x + threadIdx.x;
    if (r >= NB) return;
    float M = -1e30f, S = 0.0f;
    for (int bc = 0; bc < NUMBC; ++bc) {
        int base = (r * NUMBC + bc) * 2;
        float mb = partials[base], sb = partials[base + 1];
        if (sb > 0.0f) {
            float nM = fmaxf(M, mb);
            S = S * __expf(M - nM) + sb * __expf(mb - nM);
            M = nM;
        }
    }
    float lse = M + logf(S);
    float lp  = x_label[r] - lse;
    atomicAdd(out, -lp * (1.0f / (float)NB));
}

// ---------------------------------------------------------------------------
extern "C" void kernel_launch(void* const* d_in, const int* in_sizes, int n_in,
                              void* d_out, int out_size, void* d_ws, size_t ws_size,
                              hipStream_t stream) {
    const float* emb  = (const float*)d_in[0];   // [512, 512]
    const float* kern = (const float*)d_in[1];   // [512, 51332]
    const int*   lab  = (const int*)d_in[2];     // [512]
    float* out = (float*)d_out;

    // workspace layout (floats); Apack byte offset 211488 is 32B-aligned
    float* wsf       = (float*)d_ws;
    float* inv_norm  = wsf;                      // 51332 (padded 51336)
    float* gt_cos    = wsf + 51336;              // 512
    float* final_gt  = gt_cos + 512;             // 512
    float* x_label   = final_gt + 512;           // 512
    unsigned short* Apack = (unsigned short*)(x_label + 512);   // 512*512 bf16
    float* partials  = (float*)(Apack + NB * EMB); // 512 * 402 * 2

    arc_zero<<<1, 64, 0, stream>>>(out);
    arc_colnorm<<<(NCLS + 255) / 256, 256, 0, stream>>>(kern, inv_norm);
    arc_packA<<<(NB * EMB) / 256, 256, 0, stream>>>(emb, Apack);
    arc_gt<<<NB / 8, 256, 0, stream>>>(emb, kern, lab, inv_norm,
                                       gt_cos, final_gt, x_label);
    dim3 g(NUMBC, NB / BM);
    arc_gemm<<<g, 256, 0, stream>>>(kern, Apack, inv_norm, gt_cos, final_gt,
                                    lab, partials);
    arc_finalize<<<(NB + 255) / 256, 256, 0, stream>>>(partials, x_label, out);
}
